// SimpleMLP_76355928588795
// MI455X (gfx1250) — compile-verified, matrix operand order checked by reading, not run on hardware
//
#include <hip/hip_runtime.h>
#include <hip/hip_bf16.h>
#include <stdint.h>
#include <stddef.h>

// ---------------- CDNA5 vector types ----------------
typedef __attribute__((ext_vector_type(16))) __bf16 v16bf;
typedef __attribute__((ext_vector_type(8)))  __bf16 v8bf;
typedef __attribute__((ext_vector_type(4)))  __bf16 v4bf;
typedef __attribute__((ext_vector_type(8)))  float  v8f;

union AB16 { v16bf v; v8bf h[2]; };

// ---------------- static net description ----------------
constexpr int BATCH_  = 65536;
constexpr int DIMS_[6] = {784, 128, 32, 32, 32, 10};
constexpr int KST[5]   = {25, 4, 1, 1, 1};       // ceil(DIMS[i]/32)  (source layer)
constexpr int CTS[6]   = {0, 8, 2, 2, 2, 1};     // ceil(DIMS[o]/16)  (dest layer)
constexpr int EBASE[6] = {0, 0, 1, 3, 6, 10};    // first edge index per dest layer
constexpr int EI[15]   = {0,0,1,0,1,2,0,1,2,3,0,1,2,3,4};
constexpr int EO[15]   = {1,2,2,3,3,3,4,4,4,4,5,5,5,5,5};
// cumulative fragment / bias offsets per edge (16th entry = total)
constexpr int FRAGOFF[16] = {0,200,250,258,308,316,318,368,376,378,380,405,409,410,411,412};
constexpr int BIASOFF[16] = {0,128,160,192,224,256,288,320,352,384,416,432,448,464,480,496};
constexpr int NFRAG = 412;          // 16x16 col-tile x 32-K-step B fragments
constexpr int NBIAS = 496;          // padded bias floats

// ---------------- LDS layout (per 64-row tile) ----------------
// bf16 row strides chosen so (stride in dwords) % 8 == 4 -> conflict-free
// 16-row x b128 fragment reads across the 64 LDS banks.
constexpr int TM = 64;                                   // batch rows per block
constexpr int HS[5]     = {808, 136, 40, 40, 40};        // bf16 stride per layer
constexpr int LDSOFF[6] = {0, 103424, 120832, 125952, 131072, 136192}; // bytes
constexpr int LDS_TOTAL = 136192;

struct ParamPtrs {
    const float* w[15];
    const float* b[15];
};

// =====================================================================
// Kernel 1: pack fp32 weights/biases -> bf16 WMMA B-fragments in d_ws.
// Fragment f covers (edge e, col-tile ct, k-step ks); 512 bf16 each,
// element (lane, j):  N = ct*16 + (lane&15)
//                     K = ks*32 + (lane>>4)*8 + (j<8 ? j : j+8)
// (mirrors the documented 16-bit A-matrix lane layout; OOB -> 0)
// =====================================================================
__global__ void pack_params(ParamPtrs p, __bf16* __restrict__ packB,
                            float* __restrict__ packBias) {
    int idx = blockIdx.x * blockDim.x + threadIdx.x;
    if (idx < NFRAG * 512) {
        int f      = idx >> 9;
        int within = idx & 511;
        int lane   = within >> 4;
        int j      = within & 15;
        int e = 0;
        while (e < 14 && f >= FRAGOFF[e + 1]) ++e;
        int i = EI[e], o = EO[e];
        int nks = KST[i];
        int fl  = f - FRAGOFF[e];
        int ks  = fl % nks;
        int ct  = fl / nks;
        int n = ct * 16 + (lane & 15);
        int k = ks * 32 + (lane >> 4) * 8 + (j < 8 ? j : j + 8);
        float v = 0.0f;
        if (n < DIMS_[o] && k < DIMS_[i]) v = p.w[e][n * DIMS_[i] + k];
        packB[idx] = (__bf16)v;
    } else {
        int bi = idx - NFRAG * 512;
        if (bi < NBIAS) {
            int e = 0;
            while (e < 14 && bi >= BIASOFF[e + 1]) ++e;
            int col = bi - BIASOFF[e];
            int o   = EO[e];
            packBias[bi] = (col < DIMS_[o]) ? p.b[e][col] : 0.0f;
        }
    }
}

// =====================================================================
// Kernel 2: fused DenseNet-MLP forward. 128 threads = 4 waves; wave w
// owns batch rows [blk*64 + w*16, +16) for the entire network, so the
// only barrier is after the cooperative x-tile staging.
// =====================================================================
__global__ __launch_bounds__(128) void densemlp_fwd(
    const float* __restrict__ x,
    const __bf16* __restrict__ packB,
    const float* __restrict__ packBias,
    float* __restrict__ out) {
    __shared__ __align__(16) unsigned char smem[LDS_TOTAL];

    const int tid = threadIdx.x;

    // ---- stage x tile: 64 x 784 f32 -> bf16 LDS rows (stride 808) ----
    const size_t tileBase = (size_t)blockIdx.x * (TM * 784);
    for (int base = tid * 4; base < TM * 784; base += 128 * 4) {
        float4 v = *(const float4*)(x + tileBase + base);
        int row = base / 784;
        int col = base - row * 784;
        v4bf pv = {(__bf16)v.x, (__bf16)v.y, (__bf16)v.z, (__bf16)v.w};
        *(v4bf*)(smem + row * (HS[0] * 2) + col * 2) = pv;
    }
    // zero-fill K padding (cols 784..807 = 12 dwords per row)
    for (int z = tid; z < TM * 12; z += 128) {
        int row = z / 12;
        int dw  = z - row * 12;
        *(unsigned int*)(smem + row * (HS[0] * 2) + 784 * 2 + dw * 4) = 0u;
    }
    __syncthreads();

    const int w        = tid >> 5;        // wave id: rows w*16..w*16+15
    const int lane     = tid & 31;
    const int nIdx     = lane & 15;       // output column within tile / A row sel
    const int khalf    = lane >> 4;       // K-half for A/B fragments
    const int arowBase = w * 16 + nIdx;   // local A row this lane reads

#pragma unroll
    for (int o = 1; o <= 5; ++o) {
        for (int ct = 0; ct < CTS[o]; ++ct) {
            v8f sum = {0.f, 0.f, 0.f, 0.f, 0.f, 0.f, 0.f, 0.f};
#pragma unroll
            for (int i = 0; i < o; ++i) {
                const int e  = EBASE[o] + i;
                const float bv = packBias[BIASOFF[e] + ct * 16 + nIdx];
                v8f acc = {bv, bv, bv, bv, bv, bv, bv, bv};
                const __bf16* hrow =
                    (const __bf16*)(smem + LDSOFF[i]) + arowBase * HS[i];
                const int nks = KST[i];
                const __bf16* bbase =
                    packB + ((size_t)(FRAGOFF[e] + ct * nks) << 9) + lane * 16;
                for (int ks = 0; ks < nks; ++ks) {
                    const int k0 = ks * 32 + khalf * 8;
                    AB16 a, b;
                    a.h[0] = *(const v8bf*)(hrow + k0);
                    a.h[1] = *(const v8bf*)(hrow + k0 + 16);
                    b.h[0] = *(const v8bf*)(bbase + (size_t)ks * 512);
                    b.h[1] = *(const v8bf*)(bbase + (size_t)ks * 512 + 8);
                    acc = __builtin_amdgcn_wmma_f32_16x16x32_bf16(
                        false, a.v, false, b.v, (short)0, acc, false, false);
                }
#pragma unroll
                for (int r = 0; r < 8; ++r) sum[r] += fmaxf(acc[r], 0.0f);
            }
            if (o < 5) {
                // write h_o tile (own rows only) back to LDS as bf16
                __bf16* hd = (__bf16*)(smem + LDSOFF[o]);
#pragma unroll
                for (int r = 0; r < 8; ++r) {
                    const int row = w * 16 + khalf * 8 + r;  // C/D: M = r (+8 hi lanes)
                    hd[row * HS[o] + ct * 16 + nIdx] = (__bf16)sum[r];
                }
            } else if (nIdx < 10) {
#pragma unroll
                for (int r = 0; r < 8; ++r) {
                    const size_t row =
                        (size_t)blockIdx.x * TM + w * 16 + khalf * 8 + r;
                    out[row * 10 + nIdx] = sum[r];
                }
            }
        }
    }
}

// =====================================================================
extern "C" void kernel_launch(void* const* d_in, const int* in_sizes, int n_in,
                              void* d_out, int out_size, void* d_ws, size_t ws_size,
                              hipStream_t stream) {
    (void)in_sizes; (void)n_in; (void)out_size; (void)ws_size;
    const float* x = (const float*)d_in[0];
    ParamPtrs p;
    for (int n = 0; n < 15; ++n) {
        p.w[n] = (const float*)d_in[1 + 2 * n];
        p.b[n] = (const float*)d_in[2 + 2 * n];
    }
    __bf16* packB   = (__bf16*)d_ws;                                   // 421,888 B
    float*  packBias = (float*)((char*)d_ws + (size_t)NFRAG * 512 * 2); // + 1,984 B

    const int packTotal = NFRAG * 512 + NBIAS;
    pack_params<<<(packTotal + 255) / 256, 256, 0, stream>>>(p, packB, packBias);

    densemlp_fwd<<<BATCH_ / TM, 128, 0, stream>>>(x, packB, packBias,
                                                  (float*)d_out);
}
